// LatticeMultiHeadAttention_90494960927301
// MI455X (gfx1250) — matrix operand order repaired and checked by reading, not compiled, not run on hardware
//
#include <hip/hip_runtime.h>

#define DM    1024
#define HEADS 16
#define DKH   64
#define BATCH 2
#define SEQ   2048
#define BL    (BATCH * SEQ)

typedef __attribute__((ext_vector_type(16))) __bf16 v16bf;
typedef __attribute__((ext_vector_type(8)))  __bf16 v8bf;
typedef __attribute__((ext_vector_type(8)))  float  v8f;

__device__ __forceinline__ unsigned short f32_to_bf16(float f) {
  unsigned int u = __builtin_bit_cast(unsigned int, f);
  u += 0x7FFFu + ((u >> 16) & 1u);   // round-to-nearest-even
  return (unsigned short)(u >> 16);
}

__device__ __forceinline__ v8bf ld8(const unsigned short* p) {
  return *reinterpret_cast<const v8bf*>(p);
}

__device__ __forceinline__ v16bf cat16(v8bf lo, v8bf hi) {
  return __builtin_shufflevector(lo, hi, 0, 1, 2, 3, 4, 5, 6, 7,
                                 8, 9, 10, 11, 12, 13, 14, 15);
}

__device__ __forceinline__ v8f wmma_bf16(v16bf a, v16bf b, v8f c) {
  return __builtin_amdgcn_wmma_f32_16x16x32_bf16(false, a, false, b,
                                                 (short)0, c, false, false);
}

// Async DMA: global (row-major, no transpose) -> LDS, 16B per lane, ASYNCcnt.
__device__ __forceinline__ void async_b128_to_lds(void* lds, const void* gsrc) {
  unsigned int loff = (unsigned int)(size_t)lds;     // low 32b = LDS offset
  asm volatile("global_load_async_to_lds_b128 %0, %1, off"
               :: "v"(loff), "v"(gsrc) : "memory");
}
__device__ __forceinline__ void wait_async0() {
  asm volatile("s_wait_asynccnt 0x0" ::: "memory");
}

// ---------------------------------------------------------------------------
// 1) Sequential Kuramoto phase biases (16 heads, 4x4 lattice) — scalar.
// ---------------------------------------------------------------------------
__global__ void phase_bias_kernel(const float* __restrict__ coupling,
                                  float* __restrict__ bias) {
  if (threadIdx.x == 0 && blockIdx.x == 0) {
    float ph[HEADS];
    #pragma unroll
    for (int i = 0; i < HEADS; ++i) ph[i] = 0.f;
    const float TWO_PI = 6.283185307179586f;
    for (int h = 0; h < HEADS; ++h) {
      int r = h >> 2, c = h & 3;
      float force = 0.f;
      for (int dr = -1; dr <= 1; ++dr)
        for (int dc = -1; dc <= 1; ++dc) {
          if (dr == 0 && dc == 0) continue;
          int rr = r + dr, cc = c + dc;
          if (rr < 0 || rr > 3 || cc < 0 || cc > 3) continue;
          int j = rr * 4 + cc;
          float w = __expf(-sqrtf((float)(dr * dr + dc * dc)));
          force += w * __sinf(ph[j] - ph[h]);
        }
      float np = ph[h] + 0.01f * ((1.0f + 0.1f * (float)h) + coupling[h] * force);
      np = np - TWO_PI * floorf(np / TWO_PI);   // python % semantics
      ph[h] = np;
    }
    for (int h = 0; h < HEADS; ++h) bias[h] = 0.1f * __cosf(ph[h]);
  }
}

// ---------------------------------------------------------------------------
// 2) fp32 -> bf16 conversions
// ---------------------------------------------------------------------------
__global__ void conv_f32_to_bf16(const float* __restrict__ in,
                                 unsigned short* __restrict__ out, int n) {
  int i = blockIdx.x * blockDim.x + threadIdx.x;
  if (i < n) out[i] = f32_to_bf16(in[i]);
}

// W[h, d, k] (16,1024,64) -> W2[d, h*64 + k] (1024,1024) row-major
__global__ void conv_wproj(const float* __restrict__ W,
                           unsigned short* __restrict__ out) {
  int i = blockIdx.x * blockDim.x + threadIdx.x;
  if (i < HEADS * DM * DKH) {
    int h = i >> 16;
    int rem = i & 65535;
    int d = rem >> 6;
    int k = rem & 63;
    out[d * DM + h * DKH + k] = f32_to_bf16(W[i]);
  }
}

// ---------------------------------------------------------------------------
// 3) bf16 WMMA GEMM: C[M,N] = A[M,K] * B[K,N]; 128x128 block tile,
//    8 waves in 4(M) x 2(N), each wave 32x64 = 2x4 16x16 fragments.
//    A tile: async DMA to LDS. B tile: b128 load + transposed ds scatter.
// ---------------------------------------------------------------------------
template <bool F32OUT>
__global__ __launch_bounds__(256)
void gemm_bf16_128x128(const unsigned short* __restrict__ A,
                       const unsigned short* __restrict__ Bm,
                       unsigned short* __restrict__ Cbf,
                       float* __restrict__ Cf,
                       const float* __restrict__ bias,
                       int M, int N, int Kd) {
  __shared__ __align__(16) unsigned short Alds[128][40];  // [m][k]
  __shared__ __align__(16) unsigned short Blds[128][40];  // transposed: [n][k]
  const int tid  = threadIdx.x;
  const int lane = tid & 31, wave = tid >> 5;
  const int lh = lane & 15, ls = lane >> 4;
  const int wm = wave >> 1, wn = wave & 1;
  const int tileM = blockIdx.y * 128, tileN = blockIdx.x * 128;

  v8f acc[2][4];
  #pragma unroll
  for (int i = 0; i < 2; ++i)
    #pragma unroll
    for (int j = 0; j < 4; ++j)
      acc[i][j] = (v8f){0.f, 0.f, 0.f, 0.f, 0.f, 0.f, 0.f, 0.f};

  for (int k0 = 0; k0 < Kd; k0 += 32) {
    // A tile 128x32: 512 b128 DMAs, 2 per thread (no transpose needed)
    #pragma unroll
    for (int i = 0; i < 2; ++i) {
      int idx = i * 256 + tid;                 // 0..511
      int m = idx >> 2, kk = (idx & 3) * 8;
      async_b128_to_lds(&Alds[m][kk],
                        A + (size_t)(tileM + m) * Kd + k0 + kk);
    }
    // B tile 32x128: b128 vector load, scatter-transpose into [n][k]
    #pragma unroll
    for (int i = 0; i < 2; ++i) {
      int idx = i * 256 + tid;                 // 0..511
      int kk = idx >> 4;
      int n0 = (idx & 15) * 8;
      v8bf row = ld8(Bm + (size_t)(k0 + kk) * N + tileN + n0);
      #pragma unroll
      for (int j = 0; j < 8; ++j)
        Blds[n0 + j][kk] = __builtin_bit_cast(unsigned short, row[j]);
    }
    // prefetch next K-chunk while this one is consumed
    if (k0 + 32 < Kd) {
      __builtin_prefetch(A + (size_t)(tileM + (tid >> 1)) * Kd + k0 + 32 +
                             (tid & 1) * 16, 0, 0);
      __builtin_prefetch(Bm + (size_t)(k0 + 32 + (tid >> 3)) * N + tileN +
                             (tid & 7) * 16, 0, 0);
    }
    wait_async0();
    __syncthreads();

    v16bf af[2], bfr[4];
    #pragma unroll
    for (int fm = 0; fm < 2; ++fm) {           // A layout: K=(e<8?e:e+8)+ls*8
      int row = wm * 32 + fm * 16 + lh;
      af[fm] = cat16(ld8(&Alds[row][ls * 8]), ld8(&Alds[row][16 + ls * 8]));
    }
    #pragma unroll
    for (int fn = 0; fn < 4; ++fn) {           // B layout: K=(e<8?e:e+8)+ls*8
      int col = wn * 64 + fn * 16 + lh;
      bfr[fn] = cat16(ld8(&Blds[col][ls * 8]), ld8(&Blds[col][16 + ls * 8]));
    }
    #pragma unroll
    for (int fm = 0; fm < 2; ++fm)
      #pragma unroll
      for (int fn = 0; fn < 4; ++fn)
        acc[fm][fn] = wmma_bf16(af[fm], bfr[fn], acc[fm][fn]);
    __syncthreads();
  }

  // C layout: VGPR r -> row r + ls*8, lane -> col lh
  if (F32OUT) {
    float bv[4];
    #pragma unroll
    for (int fn = 0; fn < 4; ++fn)
      bv[fn] = bias[tileN + wn * 64 + fn * 16 + lh];
    #pragma unroll
    for (int fm = 0; fm < 2; ++fm)
      #pragma unroll
      for (int fn = 0; fn < 4; ++fn)
        #pragma unroll
        for (int r = 0; r < 8; ++r) {
          int grow = tileM + wm * 32 + fm * 16 + r + ls * 8;
          int gcol = tileN + wn * 64 + fn * 16 + lh;
          Cf[(size_t)grow * N + gcol] = acc[fm][fn][r] + bv[fn];
        }
  } else {
    #pragma unroll
    for (int fm = 0; fm < 2; ++fm)
      #pragma unroll
      for (int fn = 0; fn < 4; ++fn)
        #pragma unroll
        for (int r = 0; r < 8; ++r) {
          int grow = tileM + wm * 32 + fm * 16 + r + ls * 8;
          int gcol = tileN + wn * 64 + fn * 16 + lh;
          Cbf[(size_t)grow * N + gcol] = f32_to_bf16(acc[fm][fn][r]);
        }
  }
}

// ---------------------------------------------------------------------------
// 4) Flash attention per (b, h): 8 waves x 16 query rows = 128 rows/block,
//    streaming 32-key tiles; S and P*V on WMMA; online softmax on f32 accs.
//    K tile via async DMA; V tile via b128 load + transposed scatter.
// ---------------------------------------------------------------------------
__global__ __launch_bounds__(256)
void lattice_flash_attn(const unsigned short* __restrict__ Q,
                        const unsigned short* __restrict__ K,
                        const unsigned short* __restrict__ V,
                        const float* __restrict__ hbias,
                        unsigned short* __restrict__ O) {
  __shared__ __align__(16) unsigned short Kt[32][72];       // [key][feat]
  __shared__ __align__(16) unsigned short Vt[64][40];       // [feat][key]
  __shared__ __align__(16) unsigned short Pt[8][16][40];    // per-wave P bounce
  const int tid  = threadIdx.x;
  const int lane = tid & 31, wave = tid >> 5;
  const int lh = lane & 15, ls = lane >> 4;
  const int bb = blockIdx.y >> 4;
  const int h  = blockIdx.y & 15;
  const int qbase = blockIdx.x * 128 + wave * 16;
  const float biasH = hbias[h];
  const float scale = 0.125f;                    // 1/sqrt(64)

  // Q fragments (16 rows x 64 feats -> 2 K-chunks of 32), A layout
  const size_t qoff = ((size_t)(bb * SEQ + qbase + lh)) * DM + h * DKH;
  v16bf qf[2];
  #pragma unroll
  for (int kc = 0; kc < 2; ++kc)
    qf[kc] = cat16(ld8(Q + qoff + kc * 32 + ls * 8),
                   ld8(Q + qoff + kc * 32 + 16 + ls * 8));

  float mrow[8], lrow[8];
  v8f o[4];
  #pragma unroll
  for (int r = 0; r < 8; ++r) { mrow[r] = -1e30f; lrow[r] = 0.f; }
  #pragma unroll
  for (int f = 0; f < 4; ++f)
    o[f] = (v8f){0.f, 0.f, 0.f, 0.f, 0.f, 0.f, 0.f, 0.f};

  for (int mt = 0; mt < SEQ; mt += 32) {
    // K/V tiles 32x64: one b128 per thread each
    {
      int row = tid >> 3, c = (tid & 7) * 8;
      size_t g = ((size_t)(bb * SEQ + mt + row)) * DM + h * DKH + c;
      async_b128_to_lds(&Kt[row][c], K + g);     // natural layout -> DMA
      v8bf vv = ld8(V + g);                      // transpose via scatter
      #pragma unroll
      for (int j = 0; j < 8; ++j)
        Vt[c + j][row] = __builtin_bit_cast(unsigned short, vv[j]);
    }
    wait_async0();
    __syncthreads();

    // S = Q * K^T : B operand column n = key, K-dim = feature ((e<8?e:e+8) + ls*8)
    v8f s0 = (v8f){0.f, 0.f, 0.f, 0.f, 0.f, 0.f, 0.f, 0.f};
    v8f s1 = s0;
    #pragma unroll
    for (int kc = 0; kc < 2; ++kc) {
      v16bf b0 = cat16(ld8(&Kt[lh][kc * 32 + ls * 8]),
                       ld8(&Kt[lh][kc * 32 + 16 + ls * 8]));
      v16bf b1 = cat16(ld8(&Kt[16 + lh][kc * 32 + ls * 8]),
                       ld8(&Kt[16 + lh][kc * 32 + 16 + ls * 8]));
      s0 = wmma_bf16(qf[kc], b0, s0);
      s1 = wmma_bf16(qf[kc], b1, s1);
    }

    // Online softmax: row r+ls*8 lives in VGPR r across 16-lane group.
    #pragma unroll
    for (int r = 0; r < 8; ++r) {
      float a0 = s0[r] * scale + biasH;
      float a1 = s1[r] * scale + biasH;
      float mx = fmaxf(a0, a1);
      #pragma unroll
      for (int off = 1; off < 16; off <<= 1)
        mx = fmaxf(mx, __shfl_xor(mx, off, 32));
      float mnew  = fmaxf(mrow[r], mx);
      float alpha = __expf(mrow[r] - mnew);
      float p0 = __expf(a0 - mnew);
      float p1 = __expf(a1 - mnew);
      float rs = p0 + p1;
      #pragma unroll
      for (int off = 1; off < 16; off <<= 1)
        rs += __shfl_xor(rs, off, 32);
      lrow[r] = lrow[r] * alpha + rs;
      mrow[r] = mnew;
      #pragma unroll
      for (int f = 0; f < 4; ++f) o[f][r] *= alpha;
      Pt[wave][r + ls * 8][lh]      = f32_to_bf16(p0);   // C -> LDS
      Pt[wave][r + ls * 8][16 + lh] = f32_to_bf16(p1);
    }

    // O += P * V (P re-read in A layout; intra-wave LDS is in-order)
    v16bf pa = cat16(ld8(&Pt[wave][lh][ls * 8]),
                     ld8(&Pt[wave][lh][16 + ls * 8]));
    #pragma unroll
    for (int f = 0; f < 4; ++f) {
      v16bf vb = cat16(ld8(&Vt[f * 16 + lh][ls * 8]),
                       ld8(&Vt[f * 16 + lh][16 + ls * 8]));
      o[f] = wmma_bf16(pa, vb, o[f]);
    }
    __syncthreads();
  }

  #pragma unroll
  for (int r = 0; r < 8; ++r) {
    float inv = 1.0f / lrow[r];
    #pragma unroll
    for (int f = 0; f < 4; ++f) {
      size_t g = ((size_t)(bb * SEQ + qbase + r + ls * 8)) * DM
               + h * DKH + f * 16 + lh;
      O[g] = f32_to_bf16(o[f][r] * inv);
    }
  }
}

// ---------------------------------------------------------------------------
extern "C" void kernel_launch(void* const* d_in, const int* in_sizes, int n_in,
                              void* d_out, int out_size, void* d_ws, size_t ws_size,
                              hipStream_t stream) {
  (void)in_sizes; (void)n_in; (void)out_size; (void)ws_size;
  const float* q   = (const float*)d_in[0];
  const float* k   = (const float*)d_in[1];
  const float* v   = (const float*)d_in[2];
  const float* Wq  = (const float*)d_in[3];
  const float* Wk  = (const float*)d_in[4];
  const float* Wv  = (const float*)d_in[5];
  const float* Wo  = (const float*)d_in[6];
  const float* ob  = (const float*)d_in[7];
  const float* cpl = (const float*)d_in[8];
  float* out = (float*)d_out;

  char* ws = (char*)d_ws;
  const size_t szX = (size_t)BL * DM * 2;   // 8 MB bf16 activation buffer
  const size_t szW = (size_t)DM * DM * 2;   // 2 MB bf16 weight buffer
  unsigned short* X   = (unsigned short*)(ws);
  unsigned short* Wb  = (unsigned short*)(ws + szX);
  unsigned short* Wob = (unsigned short*)(ws + szX + szW);
  unsigned short* Qb  = (unsigned short*)(ws + szX + 2 * szW);
  unsigned short* Kb  = (unsigned short*)(ws + 2 * szX + 2 * szW);
  unsigned short* Vb  = (unsigned short*)(ws + 3 * szX + 2 * szW);
  float* biasf        = (float*)(ws + 4 * szX + 2 * szW);
  unsigned short* Ctx = X;   // X dead after V projection; reuse for ctx

  phase_bias_kernel<<<1, 32, 0, stream>>>(cpl, biasf);
  conv_f32_to_bf16<<<(DM * DM) / 256, 256, 0, stream>>>(Wo, Wob, DM * DM);

  dim3 gg(DM / 128, BL / 128);   // (8, 32)
  const int nX = BL * DM;
  const int nW = HEADS * DM * DKH;

  // Q = query @ Wq (all heads fused: 4096x1024x1024)
  conv_f32_to_bf16<<<nX / 256, 256, 0, stream>>>(q, X, nX);
  conv_wproj<<<nW / 256, 256, 0, stream>>>(Wq, Wb);
  gemm_bf16_128x128<false><<<gg, 256, 0, stream>>>(X, Wb, Qb, nullptr, nullptr, BL, DM, DM);
  // K
  conv_f32_to_bf16<<<nX / 256, 256, 0, stream>>>(k, X, nX);
  conv_wproj<<<nW / 256, 256, 0, stream>>>(Wk, Wb);
  gemm_bf16_128x128<false><<<gg, 256, 0, stream>>>(X, Wb, Kb, nullptr, nullptr, BL, DM, DM);
  // V
  conv_f32_to_bf16<<<nX / 256, 256, 0, stream>>>(v, X, nX);
  conv_wproj<<<nW / 256, 256, 0, stream>>>(Wv, Wb);
  gemm_bf16_128x128<false><<<gg, 256, 0, stream>>>(X, Wb, Vb, nullptr, nullptr, BL, DM, DM);

  // Flash attention -> ctx (bf16, heads concatenated in col = h*64+k)
  lattice_flash_attn<<<dim3(SEQ / 128, BATCH * HEADS), 256, 0, stream>>>(
      Qb, Kb, Vb, biasf, Ctx);

  // out = ctx @ out_w + out_b (fp32 output)
  gemm_bf16_128x128<true><<<gg, 256, 0, stream>>>(Ctx, Wob, nullptr, out, ob, BL, DM, DM);
}
